// TopKCrossAttention_82325933130037
// MI455X (gfx1250) — matrix operand order repaired and checked by reading, not compiled, not run on hardware
//
#include <hip/hip_runtime.h>

#define DEV __device__ __forceinline__

typedef __attribute__((ext_vector_type(16))) __bf16 v16bf;
typedef __attribute__((ext_vector_type(8)))  float  v8f;
typedef unsigned int u32x4 __attribute__((ext_vector_type(4)));
typedef int          i32x8 __attribute__((ext_vector_type(8)));
typedef int          i32x4 __attribute__((ext_vector_type(4)));

union Frag {
    uint4  q[2];   // 32 bytes = 16 bf16
    v16bf  v;
};

constexpr int   B_    = 64;
constexpr int   NQ    = 512;
constexpr int   NK    = 2048;
constexpr int   F_    = 512;
constexpr int   D_    = 128;
constexpr int   TOPK_ = 32;
constexpr float TEMP_ = 0.08838834764831845f;  // 128^-0.5
constexpr float GATE_ = 0.2f;

DEV unsigned short f2bf(float x) {
    unsigned int u = __float_as_uint(x);
    unsigned int r = (u + 0x7FFFu + ((u >> 16) & 1u)) >> 16;  // RNE
    return (unsigned short)r;
}

// ---------------------------------------------------------------------------
// TDM: async-DMA a 2D bf16 tile [rows x 32] (row stride = row_elems) from
// global memory into LDS, padded by the TDM pad feature to a 40-element
// (80-byte) LDS row stride so wave32 b128 fragment reads are conflict-free.
// D# layout per CDNA5 ISA ch.8 (groups 2/3 zero: tensor <= 2D).
// This toolchain exposes the 6-arg builtin (extra i32x8 group before cpol).
// ---------------------------------------------------------------------------
DEV void tdm_load_w_tile(unsigned lds_byte, unsigned long long gaddr,
                         int rows, int row_elems) {
    u32x4 g0;
    g0[0] = 1u;                                          // count=1 (valid), user mode
    g0[1] = lds_byte;                                    // lds_addr
    g0[2] = (unsigned)(gaddr & 0xFFFFFFFFull);           // global_addr[31:0]
    g0[3] = (unsigned)((gaddr >> 32) & 0x1FFFFFFull)     // global_addr[56:32]
          | (2u << 30);                                  // type=2 ("image")
    i32x8 g1;
    g1[0] = (1 << 16)          // data_size = 2 bytes
          | (1 << 20)          // pad_enable
          | (3 << 22)          // pad_interval: every 16 dwords (one 32-elem row)
          | (3 << 25);         // pad_amount:   4 dwords (8 bf16) -> stride 40
    g1[1] = (row_elems & 0xFFFF) << 16;                        // tensor_dim0 lo16
    g1[2] = ((row_elems >> 16) & 0xFFFF) | ((rows & 0xFFFF) << 16);  // dim0 hi / dim1 lo
    g1[3] = ((rows >> 16) & 0xFFFF) | (32 << 16);              // dim1 hi / tile_dim0=32
    g1[4] = rows & 0xFFFF;                                     // tile_dim1=rows, tile_dim2=0
    g1[5] = row_elems;                                         // tensor_dim0_stride lo32
    g1[6] = 0;                                                 // stride hi / dim1_stride
    g1[7] = 0;
    i32x4 z4 = {0, 0, 0, 0};
    i32x8 z8 = {0, 0, 0, 0, 0, 0, 0, 0};
    __builtin_amdgcn_tensor_load_to_lds(g0, g1, z4, z4, z8, 0);
}

// ---------------------------------------------------------------------------
// Kernel 0: one-shot weight conversion f32 -> bf16 (enables TDM staging).
// ---------------------------------------------------------------------------
__global__ void f32_to_bf16(const float* __restrict__ src,
                            unsigned short* __restrict__ dst, int n) {
    int i = blockIdx.x * blockDim.x + threadIdx.x;
    if (i < n) dst[i] = f2bf(src[i]);
}

// ---------------------------------------------------------------------------
// Kernel 1 (x3): Y = l2norm_row(X @ W^T) -> bf16.
// Workgroup = 32 rows x DOUT cols, 8 waves. A (activations) staged f32->bf16
// by all threads; B (weights) staged by the Tensor Data Mover, double-buffered
// so the next tile's DMA overlaps the current tile's WMMAs.
// ---------------------------------------------------------------------------
template <int FIN, int DOUT>
__global__ __launch_bounds__(256) void proj_l2norm(const float* __restrict__ X,
                                                   const unsigned short* __restrict__ Wb,
                                                   unsigned short* __restrict__ Y) {
    constexpr int CPW = DOUT / 16 / 8;  // col-tiles per wave (1 or 4)
    __shared__ unsigned short As[2][32][40];
    __shared__ unsigned short Bs[2][DOUT][40];
    __shared__ float rowsq[32];

    const int tid  = threadIdx.x;
    const int wave = tid >> 5, lane = tid & 31;
    const int m = lane & 15, hi = lane >> 4;
    const long row0 = (long)blockIdx.x * 32;

    const unsigned bs0 = (unsigned)(unsigned long long)(void*)&Bs[0][0][0];
    const unsigned bs1 = (unsigned)(unsigned long long)(void*)&Bs[1][0][0];
    const unsigned long long wbase = (unsigned long long)Wb;

    v8f acc[2][CPW];
    for (int rt = 0; rt < 2; ++rt)
        for (int j = 0; j < CPW; ++j)
            for (int e = 0; e < 8; ++e) acc[rt][j][e] = 0.0f;

    if (wave == 0) tdm_load_w_tile(bs0, wbase, DOUT, FIN);  // chunk 0 in flight

    for (int k0 = 0, it = 0; k0 < FIN; k0 += 32, ++it) {
        const int buf = it & 1;
        {   // stage A: 32x32 f32 -> bf16, 4 elems/thread
            int r  = tid >> 3;
            int kk = (tid & 7) * 4;
            float4 xv = *reinterpret_cast<const float4*>(X + (row0 + r) * FIN + k0 + kk);
            unsigned int lo  = f2bf(xv.x) | ((unsigned int)f2bf(xv.y) << 16);
            unsigned int hi2 = f2bf(xv.z) | ((unsigned int)f2bf(xv.w) << 16);
            *reinterpret_cast<uint2*>(&As[buf][r][kk]) = make_uint2(lo, hi2);
        }
        if (wave == 0) __builtin_amdgcn_s_wait_tensorcnt(0);
        __syncthreads();  // A staged, TDM tile landed, prev compute finished
        if (wave == 0 && k0 + 32 < FIN)
            tdm_load_w_tile(buf ? bs0 : bs1,
                            wbase + (unsigned long long)(k0 + 32) * 2, DOUT, FIN);

        Frag a[2];
        for (int rt = 0; rt < 2; ++rt) {  // A frag: two b128 LDS loads
            a[rt].q[0] = *reinterpret_cast<const uint4*>(&As[buf][16 * rt + m][hi * 8]);
            a[rt].q[1] = *reinterpret_cast<const uint4*>(&As[buf][16 * rt + m][16 + hi * 8]);
        }
        for (int j = 0; j < CPW; ++j) {
            int g = wave * CPW + j;
            Frag bfr;
            bfr.q[0] = *reinterpret_cast<const uint4*>(&Bs[buf][16 * g + m][hi * 8]);
            bfr.q[1] = *reinterpret_cast<const uint4*>(&Bs[buf][16 * g + m][16 + hi * 8]);
            for (int rt = 0; rt < 2; ++rt)
                acc[rt][j] = __builtin_amdgcn_wmma_f32_16x16x32_bf16(
                    false, a[rt].v, false, bfr.v, (short)0, acc[rt][j], false, false);
        }
    }
    __syncthreads();

    if (tid < 32) rowsq[tid] = 0.0f;
    __syncthreads();
    for (int rt = 0; rt < 2; ++rt)
        for (int j = 0; j < CPW; ++j)
            for (int r = 0; r < 8; ++r) {
                float v = acc[rt][j][r];
                atomicAdd(&rowsq[16 * rt + hi * 8 + r], v * v);
            }
    __syncthreads();
    for (int rt = 0; rt < 2; ++rt)
        for (int j = 0; j < CPW; ++j) {
            int g = wave * CPW + j;
            for (int r = 0; r < 8; ++r) {
                int   lr  = 16 * rt + hi * 8 + r;
                float inv = 1.0f / fmaxf(sqrtf(rowsq[lr]), 1e-12f);
                Y[(row0 + lr) * DOUT + 16 * g + m] = f2bf(acc[rt][j][r] * inv);
            }
        }
}

// ---------------------------------------------------------------------------
// Kernel 2: fused QK^T + mask + softmax-sum + top-32.
// One workgroup = (b, 16-query tile). 16 waves. Scores live in LDS only
// (16x2048 f32 = 128KB of the 320KB WGP LDS); dot_prod never touches HBM.
// ---------------------------------------------------------------------------
__global__ __launch_bounds__(512) void attn_topk(const unsigned short* __restrict__ wq,
                                                 const unsigned short* __restrict__ wk,
                                                 unsigned short* __restrict__ tvals,
                                                 int* __restrict__ tidx,
                                                 float* __restrict__ maskf) {
    __shared__ float sc[16][NK + 4];
    const int tid = threadIdx.x, wave = tid >> 5, lane = tid & 31;
    const int m = lane & 15, hi = lane >> 4;
    const int b = blockIdx.x >> 5;   // 32 q-tiles per batch
    const int qt = blockIdx.x & 31;
    const long qbase = (long)b * NQ + qt * 16;

    // A fragments (16 queries x 128 d), direct global_load_b128 from bf16 wq
    Frag a[4];
    for (int c = 0; c < 4; ++c) {
        const unsigned short* p = wq + (qbase + m) * D_ + c * 32;
        a[c].q[0] = *reinterpret_cast<const uint4*>(p + hi * 8);
        a[c].q[1] = *reinterpret_cast<const uint4*>(p + 16 + hi * 8);
    }

    const long kb = (long)b * NK;
    for (int kt = 0; kt < 8; ++kt) {  // wave handles 128 keys = 8 tiles
        int key = wave * 128 + kt * 16 + m;
        const unsigned short* kp = wk + (kb + key) * D_;
        __builtin_prefetch(kp + 16 * D_, 0, 1);  // next key tile
        v8f acc;
        for (int e = 0; e < 8; ++e) acc[e] = 0.0f;
        for (int c = 0; c < 4; ++c) {
            Frag bfr;  // B frag: w_k row is K-contiguous -> 2x b128
            bfr.q[0] = *reinterpret_cast<const uint4*>(kp + c * 32 + hi * 8);
            bfr.q[1] = *reinterpret_cast<const uint4*>(kp + c * 32 + 16 + hi * 8);
            acc = __builtin_amdgcn_wmma_f32_16x16x32_bf16(
                false, a[c].v, false, bfr.v, (short)0, acc, false, false);
        }
        int col = wave * 128 + kt * 16 + m;
        for (int r = 0; r < 8; ++r) sc[hi * 8 + r][col] = acc[r];
    }
    __syncthreads();

    // One wave per query row: max -> exp-sum -> 32x argmax (exclusion by -inf)
    const int row = wave;
    float mx = -1e30f;
    for (int i = 0; i < NK / 32; ++i) mx = fmaxf(mx, sc[row][lane + 32 * i]);
    for (int off = 16; off; off >>= 1) mx = fmaxf(mx, __shfl_xor(mx, off, 32));
    float se = 0.0f;
    for (int i = 0; i < NK / 32; ++i) se += __expf((sc[row][lane + 32 * i] - mx) * TEMP_);
    for (int off = 16; off; off >>= 1) se += __shfl_xor(se, off, 32);

    const long q = qbase + row;
    if (lane == 0) maskf[q] = (mx > GATE_) ? 1.0f : 0.0f;
    const float invs = 1.0f / se;

    for (int t = 0; t < TOPK_; ++t) {
        float best = -1e30f;
        int bidx = 0;
        for (int i = 0; i < NK / 32; ++i) {
            int idx = lane + 32 * i;
            float v = sc[row][idx];
            if (v > best) { best = v; bidx = idx; }
        }
        for (int off = 16; off; off >>= 1) {
            float ov = __shfl_xor(best, off, 32);
            int   oi = __shfl_xor(bidx, off, 32);
            if (ov > best || (ov == best && oi < bidx)) { best = ov; bidx = oi; }
        }
        if (lane == (bidx & 31)) sc[row][bidx] = -1e30f;
        if (lane == 0) {
            tvals[q * TOPK_ + t] = f2bf(__expf((best - mx) * TEMP_) * invs);
            tidx[q * TOPK_ + t]  = bidx;
        }
    }
}

// ---------------------------------------------------------------------------
// Kernel 3: per-channel gather quirk. selT[b][c][t] = w_v[b][idx[b][c][t]][c]
// (stored K-contiguous for the final WMMA's B operand).
// ---------------------------------------------------------------------------
__global__ void gather_sel(const unsigned short* __restrict__ wv,
                           const int* __restrict__ tidx,
                           unsigned short* __restrict__ selT) {
    long g = (long)blockIdx.x * blockDim.x + threadIdx.x;  // b*NQ*TOPK flat
    long bc = g >> 5;              // b*NQ + c
    int  c  = (int)(bc & (NQ - 1));
    long b  = bc >> 9;
    int  idx = tidx[g];
    selT[g] = wv[(b * NK + idx) * F_ + c];
}

// ---------------------------------------------------------------------------
// Kernel 4: gather = tvals[512x32] @ selT^T (K=32: ONE wmma per tile),
// out = gather*mask + query.
// ---------------------------------------------------------------------------
__global__ __launch_bounds__(256) void out_gemm(const unsigned short* __restrict__ tvals,
                                                const unsigned short* __restrict__ selT,
                                                const float* __restrict__ maskf,
                                                const float* __restrict__ query,
                                                float* __restrict__ out) {
    const int tid = threadIdx.x, wave = tid >> 5, lane = tid & 31;
    const int m = lane & 15, hi = lane >> 4;
    const int b  = blockIdx.x >> 4;  // 16 q-blocks of 32 rows per batch
    const int qb = blockIdx.x & 15;
    const long qbase = (long)b * NQ + qb * 32;

    Frag a[2];
    for (int rt = 0; rt < 2; ++rt) {
        const unsigned short* p = tvals + (qbase + 16 * rt + m) * TOPK_;
        a[rt].q[0] = *reinterpret_cast<const uint4*>(p + hi * 8);
        a[rt].q[1] = *reinterpret_cast<const uint4*>(p + 16 + hi * 8);
    }
    for (int j = 0; j < 4; ++j) {  // wave owns 4 col-tiles x 2 row-tiles
        int c = 16 * (wave * 4 + j) + m;
        Frag bfr;
        const unsigned short* sp = selT + ((long)b * NQ + c) * TOPK_;
        bfr.q[0] = *reinterpret_cast<const uint4*>(sp + hi * 8);
        bfr.q[1] = *reinterpret_cast<const uint4*>(sp + 16 + hi * 8);
        for (int rt = 0; rt < 2; ++rt) {
            v8f acc;
            for (int e = 0; e < 8; ++e) acc[e] = 0.0f;
            acc = __builtin_amdgcn_wmma_f32_16x16x32_bf16(
                false, a[rt].v, false, bfr.v, (short)0, acc, false, false);
            for (int r = 0; r < 8; ++r) {
                long row = qbase + 16 * rt + hi * 8 + r;
                long o = row * F_ + c;
                out[o] = acc[r] * maskf[row] + query[o];
            }
        }
    }
}

// ---------------------------------------------------------------------------
extern "C" void kernel_launch(void* const* d_in, const int* in_sizes, int n_in,
                              void* d_out, int out_size, void* d_ws, size_t ws_size,
                              hipStream_t stream) {
    const float* query    = (const float*)d_in[0];
    const float* key_feat = (const float*)d_in[1];
    const float* value    = (const float*)d_in[2];
    const float* WQ       = (const float*)d_in[3];
    const float* WK       = (const float*)d_in[4];
    const float* WV       = (const float*)d_in[5];
    float* out = (float*)d_out;

    char* ws = (char*)d_ws;
    unsigned short* wq    = (unsigned short*)(ws);                       // 8 MB
    unsigned short* wk    = (unsigned short*)(ws + (size_t)8388608);     // 32 MB
    unsigned short* wv    = (unsigned short*)(ws + (size_t)41943040);    // 128 MB
    unsigned short* tvals = (unsigned short*)(ws + (size_t)176160768);   // 2 MB
    int*            tidx  = (int*)           (ws + (size_t)178257920);   // 4 MB
    float*          maskf = (float*)         (ws + (size_t)182452224);   // 128 KB
    unsigned short* selT  = (unsigned short*)(ws + (size_t)182583296);   // 2 MB
    unsigned short* WQb   = (unsigned short*)(ws + (size_t)184680448);   // 128 KB
    unsigned short* WKb   = (unsigned short*)(ws + (size_t)184811520);   // 64 KB
    unsigned short* WVb   = (unsigned short*)(ws + (size_t)184877056);   // 256 KB

    f32_to_bf16<<<(D_ * F_)     / 256, 256, 0, stream>>>(WQ, WQb, D_ * F_);
    f32_to_bf16<<<(D_ * F_ / 2) / 256, 256, 0, stream>>>(WK, WKb, D_ * F_ / 2);
    f32_to_bf16<<<(F_ * F_ / 2) / 256, 256, 0, stream>>>(WV, WVb, F_ * F_ / 2);

    proj_l2norm<512, 128><<<(B_ * NQ) / 32, 256, 0, stream>>>(query, WQb, wq);
    proj_l2norm<256, 128><<<(B_ * NK) / 32, 256, 0, stream>>>(key_feat, WKb, wk);
    proj_l2norm<256, 512><<<(B_ * NK) / 32, 256, 0, stream>>>(value, WVb, wv);
    attn_topk<<<B_ * (NQ / 16), 512, 0, stream>>>(wq, wk, tvals, tidx, maskf);
    gather_sel<<<(B_ * NQ * TOPK_) / 256, 256, 0, stream>>>(wv, tidx, selT);
    out_gemm<<<B_ * (NQ / 32), 256, 0, stream>>>(tvals, selT, maskf, query, out);
}
